// MTGNN_55671366091496
// MI455X (gfx1250) — matrix operand order repaired and checked by reading, not compiled, not run on hardware
//
#include <hip/hip_runtime.h>
#include <hip/hip_bf16.h>
#include <math.h>

// ---- problem constants (from reference) ----
#define BN    8
#define TIN   168
#define NN    1500
#define TOUTC 24
#define CC    16
#define T2    162            // TIN - 7 + 1
#define NTILE 11             // ceil(162/16) WMMA column tiles
#define NTOPK 12
#define ALPHA_F 1.5f
#define BETA_F  0.2f

typedef __bf16 bf16;
typedef bf16  v16bf __attribute__((ext_vector_type(16)));
typedef float v8f   __attribute__((ext_vector_type(8)));
typedef unsigned int u32x4 __attribute__((ext_vector_type(4)));
typedef int i32x4 __attribute__((ext_vector_type(4)));
typedef int i32x8 __attribute__((ext_vector_type(8)));

#if defined(__has_builtin)
#  if __has_builtin(__builtin_amdgcn_tensor_load_to_lds) && __has_builtin(__builtin_amdgcn_s_wait_tensorcnt)
#    define HAVE_TDM 1
#  else
#    define HAVE_TDM 0
#  endif
#else
#  define HAVE_TDM 0
#endif

// K-index held by element e (0..15) of the 16-bit A-operand (16x32) for `lane`
// contiguous order (matches B fills): lanes 0-15 hold K {0..15}, lanes 16-31 {16..31}.
__device__ __forceinline__ int a_kmap(int lane, int e) {
  int kb   = (lane < 16) ? 0 : 16;
  return kb + e;
}

// ------------------------------------------------------------------
// K0: fold pointwise conv + 4-branch inception into one 16x7 filter,
//     and precompute the zero-column constant k0 for the fused head.
// ------------------------------------------------------------------
__global__ __launch_bounds__(32) void k_setup(
    const float* pw, const float* pb,
    const float* tw2, const float* tb2, const float* tw3, const float* tb3,
    const float* tw6, const float* tb6, const float* tw7, const float* tb7,
    const float* mb, const float* rw1, const float* rb1, const float* rw2,
    float* weff, float* tbp, float* k0) {
  int t = threadIdx.x;
  if (t < 16) {
    int co = t, br = co >> 2, o = co & 3;
    const float* tw; const float* tb; int kk;
    if (br == 0)      { tw = tw2; tb = tb2; kk = 2; }
    else if (br == 1) { tw = tw3; tb = tb3; kk = 3; }
    else if (br == 2) { tw = tw6; tb = tb6; kk = 6; }
    else              { tw = tw7; tb = tb7; kk = 7; }
    for (int j = 0; j < 8; ++j) weff[co * 8 + j] = 0.0f;
    for (int j = 7 - kk; j < 7; ++j) {
      float s = 0.0f;
      for (int cin = 0; cin < 16; ++cin)
        s += tw[(o * 16 + cin) * kk + (j - (7 - kk))] * pw[cin];
      weff[co * 8 + j] = s;
    }
    float bb = tb[o];
    for (int cin = 0; cin < 16; ++cin) {
      float sj = 0.0f;
      for (int jj = 0; jj < kk; ++jj) sj += tw[(o * 16 + cin) * kk + jj];
      bb += pb[cin] * sj;
    }
    tbp[co] = bb;
  }
  if (t == 16) {  // k0 = rw2 @ relu(rw1 @ relu(mb) + rb1)   (no rb2)
    float z1[16], z2[16];
    for (int c = 0; c < 16; ++c) z1[c] = fmaxf(mb[c], 0.0f);
    for (int c = 0; c < 16; ++c) {
      float s = rb1[c];
      for (int k = 0; k < 16; ++k) s += rw1[c * 16 + k] * z1[k];
      z2[c] = fmaxf(s, 0.0f);
    }
    for (int o = 0; o < TOUTC; ++o) {
      float s = 0.0f;
      for (int c = 0; c < 16; ++c) s += rw2[o * 16 + c] * z2[c];
      k0[o] = s;
    }
  }
}

// ------------------------------------------------------------------
// K1: m1 = tanh(a*(e1@gw1+gb1)), m2 = tanh(a*(e2@gw2+gb2))
// ------------------------------------------------------------------
__global__ void k_embed(const float* e1, const float* e2,
                        const float* gw1, const float* gb1,
                        const float* gw2, const float* gb2,
                        float* m1, float* m2) {
  int id = blockIdx.x * blockDim.x + threadIdx.x;
  if (id >= 2 * NN * 16) return;
  int which = id / (NN * 16);
  int r = id - which * (NN * 16);
  int n = r >> 4, e = r & 15;
  const float* emb = which ? e2 : e1;
  const float* gw  = which ? gw2 : gw1;
  const float* gb  = which ? gb2 : gb1;
  float s = gb[e];
  for (int k = 0; k < 16; ++k) s += emb[n * 16 + k] * gw[k * 16 + e];
  (which ? m2 : m1)[n * 16 + e] = tanhf(ALPHA_F * s);
}

// ------------------------------------------------------------------
// K2: per-row scores + top-12 + build row-normalized sparse (A+I) rows
//     stored as 13 (idx,weight) pairs (entry 0 = self).
// ------------------------------------------------------------------
__global__ void k_graph(const float* m1, const float* m2, int* aidx, float* aw) {
  int v = blockIdx.x * blockDim.x + threadIdx.x;
  if (v >= NN) return;
  float m1v[16], m2v[16];
  for (int k = 0; k < 16; ++k) { m1v[k] = m1[v * 16 + k]; m2v[k] = m2[v * 16 + k]; }
  float tv[NTOPK]; int ti[NTOPK];
  for (int i = 0; i < NTOPK; ++i) { tv[i] = -1.0f; ti[i] = 0; }
  for (int w = 0; w < NN; ++w) {
    float s1 = 0.0f, s2 = 0.0f;
    for (int k = 0; k < 16; ++k) {
      s1 += m1v[k] * m2[w * 16 + k];
      s2 += m2v[k] * m1[w * 16 + k];
    }
    float a = tanhf(ALPHA_F * (s1 - s2));
    a = fmaxf(a, 0.0f);
    if (a > tv[NTOPK - 1]) {          // strict > keeps lower index on ties (lax.top_k)
      int p = NTOPK - 1;
      while (p > 0 && tv[p - 1] < a) { tv[p] = tv[p - 1]; ti[p] = ti[p - 1]; --p; }
      tv[p] = a; ti[p] = w;
    }
  }
  float sum = 1.0f;
  for (int i = 0; i < NTOPK; ++i) sum += tv[i];
  float inv = 1.0f / sum;
  aidx[v * 16 + 0] = v; aw[v * 16 + 0] = inv;          // self loop (the +I)
  for (int i = 0; i < NTOPK; ++i) { aidx[v * 16 + 1 + i] = ti[i]; aw[v * 16 + 1 + i] = tv[i] * inv; }
  for (int i = NTOPK + 1; i < 16; ++i) { aidx[v * 16 + i] = v; aw[v * 16 + i] = 0.0f; }
}

// ------------------------------------------------------------------
// K3: fused front-end: h0[b,c,n,t] = relu(tbp[c] + sum_j weff[c][j]*x[b,t+j,n])
//     written as bf16 into the hop-0 slab of ho[b][hop*16+c][n][t].
// ------------------------------------------------------------------
__global__ __launch_bounds__(256) void k_front(const float* x, const float* weff,
                                               const float* tbp, bf16* ho) {
  __shared__ float xs[TIN];
  __shared__ float wsh[128];
  __shared__ float bsh[16];
  int b = blockIdx.x / NN, n = blockIdx.x % NN;
  int tid = threadIdx.x;
  if (tid < TIN) xs[tid] = x[((size_t)b * TIN + tid) * NN + n];
  if (tid < 128) wsh[tid] = weff[tid];
  if (tid >= 128 && tid < 144) bsh[tid - 128] = tbp[tid - 128];
  __syncthreads();
  for (int e = tid; e < 16 * T2; e += 256) {
    int c = e / T2, t = e - c * T2;
    float s = bsh[c];
#pragma unroll
    for (int j = 0; j < 7; ++j) s += wsh[c * 8 + j] * xs[t + j];
    ho[(((size_t)b * 48 + c) * NN + n) * T2 + t] = (bf16)fmaxf(s, 0.0f);
  }
}

// ------------------------------------------------------------------
// K4: one mix-hop step: ho[hop] = beta*ho[0] + (1-beta)*A_sparse @ ho[hop-1]
//     13-nnz gather per row; hop slabs (62 MB) are L2-resident (192 MB L2).
// ------------------------------------------------------------------
__global__ __launch_bounds__(192) void k_hop(bf16* ho, const int* aidx, const float* aw, int hop) {
  int idx = blockIdx.x;
  int v = idx % NN;
  int c = (idx / NN) & 15;
  int b = idx / (NN * 16);
  __shared__ int   si[16];
  __shared__ float sw[16];
  if (threadIdx.x < 13) { si[threadIdx.x] = aidx[v * 16 + threadIdx.x]; sw[threadIdx.x] = aw[v * 16 + threadIdx.x]; }
  __syncthreads();
  int t = threadIdx.x;
  if (t >= T2) return;
  float hin = (float)ho[(((size_t)b * 48 + c) * NN + v) * T2 + t];
  size_t srcc = ((size_t)b * 48 + (hop - 1) * 16 + c) * NN;
  float s = 0.0f;
#pragma unroll
  for (int j = 0; j < 13; ++j) s += sw[j] * (float)ho[(srcc + (size_t)si[j]) * T2 + t];
  ho[(((size_t)b * 48 + hop * 16 + c) * NN + v) * T2 + t] = (bf16)(BETA_F * hin + (1.0f - BETA_F) * s);
}

// ------------------------------------------------------------------
// TDM: DMA one 48-row x 162-col bf16 tile (row stride NN*T2 elems) into LDS.
// D# packing per CDNA5 ISA §8.3/8.4 (group0: count/lds/global/type,
// group1: data_size=2B, tensor 162x48, tile 162x48, dim0 stride NN*T2).
// ------------------------------------------------------------------
#if HAVE_TDM
__device__ __forceinline__ void tdm_load_tile(const bf16* gsrc, unsigned lds_off) {
  unsigned long long ga = (unsigned long long)(const void*)gsrc;
  u32x4 g0;
  g0[0] = 1u;                                              // count=1 (valid user D#)
  g0[1] = lds_off;                                         // lds_addr (bytes)
  g0[2] = (unsigned)ga;                                    // global_addr[31:0]
  g0[3] = (unsigned)((ga >> 32) & 0x01FFFFFFu) | (2u << 30); // addr[56:32] | type=2
  unsigned long long s0 = (unsigned long long)NN * T2;     // row stride (data_size units)
  i32x8 g1;
  g1[0] = (int)(1u << 16);                                 // data_size = 2 bytes
  g1[1] = (int)(((unsigned)T2 & 0xFFFFu) << 16);           // tensor_dim0[15:0]
  g1[2] = (int)(((unsigned)T2 >> 16) | (48u << 16));       // tensor_dim0 hi | tensor_dim1 lo
  g1[3] = (int)((48u >> 16) | ((unsigned)T2 << 16));       // tensor_dim1 hi | tile_dim0=162
  g1[4] = (int)48u;                                        // tile_dim1=48, tile_dim2=0
  g1[5] = (int)(unsigned)s0;                               // tensor_dim0_stride[31:0]
  g1[6] = (int)(unsigned)(s0 >> 32);                       // stride hi | dim1_stride lo (0)
  g1[7] = 0;
  i32x4 gz = {0, 0, 0, 0};
#if __clang_major__ >= 23
  i32x8 g5 = {0, 0, 0, 0, 0, 0, 0, 0};
  __builtin_amdgcn_tensor_load_to_lds(g0, g1, gz, gz, g5, 0);
#else
  __builtin_amdgcn_tensor_load_to_lds(g0, g1, gz, gz, 0);
#endif
}
#endif

// ------------------------------------------------------------------
// K5: fused WMMA head. One wave per (b,n): 48x162 bf16 tile in LDS via TDM,
//     chain mw(16x48) -> relu -> rw1(16x16) -> relu -> rw2(24x16),
//     time-mean accumulated in the WMMA C accumulator across 11 tiles.
// ------------------------------------------------------------------
__global__ __launch_bounds__(128) void k_head(const bf16* __restrict__ ho,
    const float* mw, const float* mb, const float* rw1, const float* rb1,
    const float* rw2, const float* rb2, const float* k0, float* out) {
  __shared__ bf16 S[4][48 * T2];
  const int wid  = threadIdx.x >> 5;
  const int lane = threadIdx.x & 31;
  const int bn   = blockIdx.x * 4 + wid;
  const int b = bn / NN, n = bn % NN;
  const int col = lane & 15;
  const int hi  = lane >> 4;          // 0: rows 0-7 / K 0-15 half; 1: rows 8-15 / K 16-31 half
  bf16* Sw = &S[wid][0];
  const bf16* gsrc = ho + (((size_t)b * 48 + 0) * NN + n) * T2;

#if HAVE_TDM
  // async tensor DMA of this wave's tile; LDS offset = low 32 bits of flat addr
  tdm_load_tile(gsrc, (unsigned)(unsigned long long)(uintptr_t)Sw);
#else
  for (int k = 0; k < 48; ++k) {
    const bf16* src = gsrc + (size_t)k * NN * T2;
    for (int t0 = 0; t0 < T2; t0 += 64) {
      int tt = t0 + lane * 2;
      if (tt < T2) *(unsigned int*)(Sw + k * T2 + tt) = *(const unsigned int*)(src + tt);
    }
  }
#endif

  // pack weight A-operands while the DMA is in flight (row = out channel = col)
  v16bf Amw0, Amw1, Arw1, Arw2lo, Arw2hi;
  const int row = col;
#pragma unroll
  for (int e = 0; e < 16; ++e) {
    int km = a_kmap(lane, e);                    // 0..31
    Amw0[e] = (bf16)mw[row * 48 + km];
    int k1 = 32 + km;
    Amw1[e]   = (bf16)((k1 < 48) ? mw[row * 48 + k1] : 0.0f);
    Arw1[e]   = (bf16)((km < 16) ? rw1[row * 16 + km] : 0.0f);
    Arw2lo[e] = (bf16)((km < 16) ? rw2[row * 16 + km] : 0.0f);
    Arw2hi[e] = (bf16)((km < 16 && row < 8) ? rw2[(16 + row) * 16 + km] : 0.0f);
  }

#if HAVE_TDM
  __builtin_amdgcn_s_wait_tensorcnt(0);
#else
  __syncthreads();
#endif

  v8f c3 = {}, c4 = {};                          // rw2 accumulators (sum over t)
  for (int tt = 0; tt < NTILE; ++tt) {
    const int tcol = tt * 16 + col;
    // B-operand: K = hi*16 + e, column = col (contiguous K order, same as a_kmap);
    // columns >= 162 (tile 10 pad) read as exact zeros.
    v16bf B0, B1;
#pragma unroll
    for (int e = 0; e < 16; ++e) {
      int k  = hi * 16 + e;
      B0[e] = (tcol < T2) ? Sw[k * T2 + tcol] : (bf16)0.0f;
      int kb = 32 + k;
      B1[e] = (kb < 48 && tcol < T2) ? Sw[kb * T2 + tcol] : (bf16)0.0f;
    }
    v8f c1;
#pragma unroll
    for (int r = 0; r < 8; ++r) c1[r] = mb[r + hi * 8];
    c1 = __builtin_amdgcn_wmma_f32_16x16x32_bf16(false, Amw0, false, B0, (short)0, c1, false, false);
    c1 = __builtin_amdgcn_wmma_f32_16x16x32_bf16(false, Amw1, false, B1, (short)0, c1, false, false);
#pragma unroll
    for (int r = 0; r < 8; ++r) c1[r] = fmaxf(c1[r], 0.0f);

    // repack D (16x16 f32) into bf16 B-operand: rows 8-15 live in partner lane +16
    v16bf B2;
#pragma unroll
    for (int e = 0; e < 8; ++e) {
      float own = c1[e];
      float oth = __shfl(c1[e], col + 16);
      B2[e]     = (bf16)(hi ? 0.0f : own);
      B2[e + 8] = (bf16)(hi ? 0.0f : oth);
    }
    v8f c2;
#pragma unroll
    for (int r = 0; r < 8; ++r) c2[r] = rb1[r + hi * 8];
    c2 = __builtin_amdgcn_wmma_f32_16x16x32_bf16(false, Arw1, false, B2, (short)0, c2, false, false);
#pragma unroll
    for (int r = 0; r < 8; ++r) c2[r] = fmaxf(c2[r], 0.0f);

    v16bf B3;
#pragma unroll
    for (int e = 0; e < 8; ++e) {
      float own = c2[e];
      float oth = __shfl(c2[e], col + 16);
      B3[e]     = (bf16)(hi ? 0.0f : own);
      B3[e + 8] = (bf16)(hi ? 0.0f : oth);
    }
    c3 = __builtin_amdgcn_wmma_f32_16x16x32_bf16(false, Arw2lo, false, B3, (short)0, c3, false, false);
    c4 = __builtin_amdgcn_wmma_f32_16x16x32_bf16(false, Arw2hi, false, B3, (short)0, c4, false, false);
  }

  // remove the constant contribution of the 14 zero-padded columns (tile 10, col>=2)
  if (col >= 2) {
#pragma unroll
    for (int r = 0; r < 8; ++r) {
      c3[r] -= k0[r + hi * 8];
      if (!hi) c4[r] -= k0[16 + r];
    }
  }
  // sum over the 16 columns held by each half-wave
  for (int off = 1; off < 16; off <<= 1) {
#pragma unroll
    for (int r = 0; r < 8; ++r) {
      c3[r] += __shfl_xor(c3[r], off);
      c4[r] += __shfl_xor(c4[r], off);
    }
  }
  if (col == 0) {
    const float inv = 1.0f / (float)T2;
#pragma unroll
    for (int r = 0; r < 8; ++r) {
      int M = r + hi * 8;                                   // rows 0..15
      out[((size_t)b * TOUTC + M) * NN + n] = c3[r] * inv + rb2[M];
      if (!hi) {
        int M4 = 16 + r;                                    // rows 16..23
        out[((size_t)b * TOUTC + M4) * NN + n] = c4[r] * inv + rb2[M4];
      }
    }
  }
}

// ------------------------------------------------------------------
extern "C" void kernel_launch(void* const* d_in, const int* in_sizes, int n_in,
                              void* d_out, int out_size, void* d_ws, size_t ws_size,
                              hipStream_t stream) {
  (void)in_sizes; (void)n_in; (void)out_size; (void)ws_size;
  const float* x   = (const float*)d_in[0];
  const float* pw  = (const float*)d_in[1];
  const float* pb  = (const float*)d_in[2];
  const float* tw2 = (const float*)d_in[3];
  const float* tb2 = (const float*)d_in[4];
  const float* tw3 = (const float*)d_in[5];
  const float* tb3 = (const float*)d_in[6];
  const float* tw6 = (const float*)d_in[7];
  const float* tb6 = (const float*)d_in[8];
  const float* tw7 = (const float*)d_in[9];
  const float* tb7 = (const float*)d_in[10];
  const float* e1  = (const float*)d_in[11];
  const float* e2  = (const float*)d_in[12];
  const float* gw1 = (const float*)d_in[13];
  const float* gb1 = (const float*)d_in[14];
  const float* gw2 = (const float*)d_in[15];
  const float* gb2 = (const float*)d_in[16];
  const float* mw  = (const float*)d_in[17];
  const float* mb  = (const float*)d_in[18];
  const float* rw1 = (const float*)d_in[19];
  const float* rb1 = (const float*)d_in[20];
  const float* rw2 = (const float*)d_in[21];
  const float* rb2 = (const float*)d_in[22];
  float* out = (float*)d_out;

  char*  ws   = (char*)d_ws;
  float* weff = (float*)ws;                 // 128 f32
  float* tbp  = weff + 128;                 // 16
  float* k0   = tbp + 16;                   // 24 (pad)
  float* m1   = (float*)(ws + 1024);        // 24000 f32
  float* m2   = m1 + NN * 16;               // 24000 f32
  int*   aidx = (int*)(m2 + NN * 16);       // 24000 i32
  float* aw   = (float*)(aidx + NN * 16);   // 24000 f32
  bf16*  ho   = (bf16*)(ws + 385024);       // B*48*N*T2 bf16 = 186.6 MB

  k_setup<<<1, 32, 0, stream>>>(pw, pb, tw2, tb2, tw3, tb3, tw6, tb6, tw7, tb7,
                                mb, rw1, rb1, rw2, weff, tbp, k0);
  k_embed<<<(2 * NN * 16 + 255) / 256, 256, 0, stream>>>(e1, e2, gw1, gb1, gw2, gb2, m1, m2);
  k_graph<<<(NN + 63) / 64, 64, 0, stream>>>(m1, m2, aidx, aw);
  k_front<<<BN * NN, 256, 0, stream>>>(x, weff, tbp, ho);
  k_hop<<<BN * 16 * NN, 192, 0, stream>>>(ho, aidx, aw, 1);
  k_hop<<<BN * 16 * NN, 192, 0, stream>>>(ho, aidx, aw, 2);
  k_head<<<(BN * NN) / 4, 128, 0, stream>>>(ho, mw, mb, rw1, rb1, rw2, rb2, k0, out);
}
